// AECNF_635655160406
// MI455X (gfx1250) — matrix operand order repaired and checked by reading, not compiled, not run on hardware
//
#include <hip/hip_runtime.h>
#include <math.h>

// ---------------------------------------------------------------------------
// CDNA5 (gfx1250) fused implementation of the VAE + CNF reference.
// All large GEMMs run through v_wmma_f32_16x16x32_f16 (f16 in, f32 acc).
// GEMM inner loops are completely branch-free (weights are row- and
// column-padded with zeros), so accumulators stay pinned in registers.
// ---------------------------------------------------------------------------

typedef _Float16 f16;
typedef _Float16 v16h __attribute__((ext_vector_type(16)));
typedef _Float16 v8h  __attribute__((ext_vector_type(8)));
typedef float    v8f  __attribute__((ext_vector_type(8)));

#define B_SZ   8192
#define D_IN   784
#define KP_IN  800     // 784 padded to multiple of 32
#define HID    512
#define LD     16
#define CD     32
#define OH     256
#define WID    256
#define NSTEP  16

// ---------------- fragment loaders (row-major, K contiguous) ----------------
// A (16x32 f16): lane holds row (m0+lane%16); K chunks at k0+half*8 and k0+16+half*8
__device__ __forceinline__ v16h load_a_frag(const f16* row, int k0, int half) {
  v8h c0 = *(const v8h*)(row + k0 + half * 8);
  v8h c1 = *(const v8h*)(row + k0 + 16 + half * 8);
  v16h a;
#pragma unroll
  for (int e = 0; e < 8; ++e) { a[e] = c0[e]; a[8 + e] = c1[e]; }
  return a;
}
// B (32x16 f16, fed as N x K row-major "W"): lane holds col (n0+lane%16);
// K chunks at k0+half*8 and k0+16+half*8
__device__ __forceinline__ v16h load_b_frag(const f16* row, int k0, int half) {
  v8h c0 = *(const v8h*)(row + k0 + half * 8);
  v8h c1 = *(const v8h*)(row + k0 + 16 + half * 8);
  v16h b;
#pragma unroll
  for (int e = 0; e < 8; ++e) { b[e] = c0[e]; b[8 + e] = c1[e]; }
  return b;
}

__device__ __forceinline__ float apply_act(float v, int act) {
  if (act == 1) return v > 0.f ? v : expf(v) - 1.f;      // ELU
  if (act == 2) return tanhf(v);                          // tanh
  if (act == 3) return 1.f / (1.f + expf(-v));            // sigmoid
  return v;
}

// ---------------- f32 -> f16 with row and column (K) zero padding -----------
__global__ void convert_pad_kernel(const float* __restrict__ src, f16* __restrict__ dst,
                                   int rows, int rowsPad, int K, int Kp) {
  long long idx = (long long)blockIdx.x * 256 + threadIdx.x;
  long long total = (long long)rowsPad * Kp;
  if (idx >= total) return;
  int r = (int)(idx / Kp), k = (int)(idx % Kp);
  f16 v = (f16)0.f;
  if (r < rows && k < K) v = (f16)src[(size_t)r * K + k];
  dst[idx] = v;
}

// ---------------- generic WMMA NT GEMM: C = act(A * W^T + bias) -------------
// A: [M x Kp] f16 row-major, W: [Npad x Kp] f16 row-major, both zero-padded.
// grid.x = M/32, grid.y covers N in 512-wide chunks; 8 waves/block.
// Each wave computes a 32(M) x 64(N) C macro-tile (2x4 WMMA tiles).
// The K loop body has NO control flow: garbage tiles hit zero weight rows
// and are dropped in the store epilogue.
__global__ __launch_bounds__(256) void gemm_nt_kernel(
    const f16* __restrict__ A, const f16* __restrict__ W,
    const float* __restrict__ bias, int Kp, int N, int Npad, int act,
    f16* __restrict__ C16, int ldc16, float* __restrict__ C32, int ldc32) {
  const int lane = threadIdx.x & 31;
  // readfirstlane -> wave id is an SGPR: tile guards are scalar branches and
  // EXEC is never modified around the WMMA ops.
  const int wid  = __builtin_amdgcn_readfirstlane(threadIdx.x >> 5);
  const int lr   = lane & 15;
  const int half = lane >> 4;
  const int m0 = blockIdx.x * 32;
  const int n0 = blockIdx.y * 512 + wid * 64;
  if (n0 >= Npad) return;   // scalar early-exit, before any accumulator

  const f16* Arow0 = A + (size_t)(m0 + lr) * Kp;
  const f16* Arow1 = A + (size_t)(m0 + 16 + lr) * Kp;
  const f16* Wrow0 = W + (size_t)(n0 + lr) * Kp;
  const f16* Wrow1 = W + (size_t)(n0 + 16 + lr) * Kp;
  const f16* Wrow2 = W + (size_t)(n0 + 32 + lr) * Kp;
  const f16* Wrow3 = W + (size_t)(n0 + 48 + lr) * Kp;

  v8f a00 = {}, a01 = {}, a02 = {}, a03 = {};
  v8f a10 = {}, a11 = {}, a12 = {}, a13 = {};
  for (int k0 = 0; k0 < Kp; k0 += 32) {
    v16h af0 = load_a_frag(Arow0, k0, half);
    v16h af1 = load_a_frag(Arow1, k0, half);
    v16h b0 = load_b_frag(Wrow0, k0, half);
    v16h b1 = load_b_frag(Wrow1, k0, half);
    v16h b2 = load_b_frag(Wrow2, k0, half);
    v16h b3 = load_b_frag(Wrow3, k0, half);
    a00 = __builtin_amdgcn_wmma_f32_16x16x32_f16(false, af0, false, b0, (short)0, a00, false, false);
    a10 = __builtin_amdgcn_wmma_f32_16x16x32_f16(false, af1, false, b0, (short)0, a10, false, false);
    a01 = __builtin_amdgcn_wmma_f32_16x16x32_f16(false, af0, false, b1, (short)0, a01, false, false);
    a11 = __builtin_amdgcn_wmma_f32_16x16x32_f16(false, af1, false, b1, (short)0, a11, false, false);
    a02 = __builtin_amdgcn_wmma_f32_16x16x32_f16(false, af0, false, b2, (short)0, a02, false, false);
    a12 = __builtin_amdgcn_wmma_f32_16x16x32_f16(false, af1, false, b2, (short)0, a12, false, false);
    a03 = __builtin_amdgcn_wmma_f32_16x16x32_f16(false, af0, false, b3, (short)0, a03, false, false);
    a13 = __builtin_amdgcn_wmma_f32_16x16x32_f16(false, af1, false, b3, (short)0, a13, false, false);
  }

  v8f acc0[4] = {a00, a01, a02, a03};
  v8f acc1[4] = {a10, a11, a12, a13};
#pragma unroll
  for (int j = 0; j < 4; ++j) {
    const int nt = n0 + j * 16;
    if (nt >= N) continue;                     // scalar, epilogue only
    const int n = nt + lr;
    const float bn = bias[n];
#pragma unroll
    for (int r = 0; r < 8; ++r) {
      const int m = m0 + r + 8 * half;
      float v0 = apply_act(acc0[j][r] + bn, act);
      float v1 = apply_act(acc1[j][r] + bn, act);
      if (C16) {
        C16[(size_t)m * ldc16 + n] = (f16)v0;
        C16[(size_t)(m + 16) * ldc16 + n] = (f16)v1;
      }
      if (C32) {
        C32[(size_t)m * ldc32 + n] = v0;
        C32[(size_t)(m + 16) * ldc32 + n] = v1;
      }
    }
  }
}

// ---------------- latent: mean/std/z from ms ----------------
__global__ void latent_kernel(const float* __restrict__ ms, const float* __restrict__ eps,
                              float* __restrict__ o_mean, float* __restrict__ o_std,
                              float* __restrict__ z32, f16* __restrict__ z16) {
  int idx = blockIdx.x * 256 + threadIdx.x;
  if (idx >= B_SZ * LD) return;
  int b = idx >> 4, j = idx & 15;
  float mean = ms[b * 32 + j];
  float logs = ms[b * 32 + 16 + j];
  float sd = expf(logs) + 1e-6f;
  float z = mean + eps[idx] * sd;
  o_mean[idx] = mean;
  o_std[idx] = sd;
  z32[idx] = z;
  z16[b * 32 + j] = (f16)z;        // K padded 16 -> 32
  z16[b * 32 + 16 + j] = (f16)0.f;
}

// ---------------- ccontrib = emb[cond] @ cwc^T ----------------
__global__ void ccond_kernel(const float* __restrict__ emb, const int* __restrict__ cond,
                             const float* __restrict__ cwc, float* __restrict__ cc) {
  int idx = blockIdx.x * 256 + threadIdx.x;
  if (idx >= B_SZ * 2 * LD) return;
  int b = idx >> 5, j = idx & 31;
  const float* e = emb + cond[b] * CD;
  float s = 0.f;
#pragma unroll
  for (int i = 0; i < CD; ++i) s += e[i] * cwc[j * CD + i];
  cc[idx] = s;
}

// ---------------- hypernet: 48 eval-time parameter sets ----------------
// set = step*3 + {0:t, 1:t+dt/2, 2:t+dt}
// Emits pre-padded operands so the CNF fragments are pure vector loads:
//   WmP : [256][32] f16, K 16..31 zeroed
//   UmT : [16][256] f16 (Um transposed)
//   wuP : [16][256] f16, rows 1..15 zeroed
//   Bm  : [256] f32
__global__ __launch_bounds__(256) void hyper_kernel(
    const float* __restrict__ hw1, const float* __restrict__ hb1,
    const float* __restrict__ hw2, const float* __restrict__ hb2,
    const float* __restrict__ hw3, const float* __restrict__ hb3,
    f16* __restrict__ WmP16, f16* __restrict__ UmT16,
    f16* __restrict__ wuP16, float* __restrict__ Bm32) {
  __shared__ float x1[OH], x2[OH];
  __shared__ float WmL[WID * LD];
  __shared__ float UmL[WID * LD];
  int t = threadIdx.x, set = blockIdx.x;
  int step = set / 3, which = set % 3;
  const float dt = -0.625f;
  float tt = 10.f + dt * step + (which == 0 ? 0.f : (which == 1 ? dt * 0.5f : dt));
  x1[t] = tanhf(tt * hw1[t] + hb1[t]);
  __syncthreads();
  float a2 = hb2[t];
  for (int j = 0; j < OH; ++j) a2 += x1[j] * hw2[t * OH + j];
  x2[t] = tanhf(a2);
  __syncthreads();

  f16* WmS  = WmP16 + (size_t)set * WID * 32;
  f16* UmTS = UmT16 + (size_t)set * WID * LD;
  f16* wuS  = wuP16 + (size_t)set * LD * WID;
  float* BmS = Bm32 + (size_t)set * WID;

  for (int k = 0; k < 16; ++k) {           // Wm rows: p[0..4096)
    int r = t + 256 * k;
    float p = hb3[r];
    for (int j = 0; j < OH; ++j) p += x2[j] * hw3[(size_t)r * OH + j];
    WmL[r] = p;
    int n = r >> 4, jc = r & 15;
    WmS[n * 32 + jc] = (f16)p;             // [256][32], K padded
    WmS[n * 32 + 16 + jc] = (f16)0.f;
  }
  for (int k = 0; k < 16; ++k) {           // Um = p[4096..8192) * sigmoid(p[8192..12288))
    int i = t + 256 * k;
    int r1 = WID * LD + i, r2 = 2 * WID * LD + i;
    float p1 = hb3[r1], p2 = hb3[r2];
    for (int j = 0; j < OH; ++j) {
      p1 += x2[j] * hw3[(size_t)r1 * OH + j];
      p2 += x2[j] * hw3[(size_t)r2 * OH + j];
    }
    float um = p1 * (1.f / (1.f + expf(-p2)));
    UmL[i] = um;
    int kk = i >> 4, jc = i & 15;          // Um[kk][jc]
    UmTS[jc * WID + kk] = (f16)um;         // transposed [16][256] for WMMA B
  }
  {                                        // Bm: p[12288..12544)
    int r = 3 * WID * LD + t;
    float p = hb3[r];
    for (int j = 0; j < OH; ++j) p += x2[j] * hw3[(size_t)r * OH + j];
    BmS[t] = p;
  }
  __syncthreads();
  float s = 0.f;                           // wu[n] = sum_j Wm[n][j]*Um[n][j]
#pragma unroll
  for (int j = 0; j < LD; ++j) s += WmL[t * LD + j] * UmL[t * LD + j];
  wuS[t] = (f16)s;                         // row 0 = wu
#pragma unroll
  for (int rr = 1; rr < 16; ++rr) wuS[rr * WID + t] = (f16)0.f;  // rows 1..15 = 0
}

// ---------------- fused CNF RK4 integrator ----------------
// 4 waves/block, each wave owns 16 samples; WMMA for hh, dz and trace.
// All fragments are plain vector loads (operands pre-padded by hyper_kernel),
// so there is no select/branch anywhere near the WMMA ops.
__global__ __launch_bounds__(128) void cnf_kernel(
    const float* __restrict__ zin, const float* __restrict__ cc,
    const float* __restrict__ cwz, const float* __restrict__ cbz,
    const float* __restrict__ cbc,
    const f16* __restrict__ hypWmP, const f16* __restrict__ hypUmT,
    const f16* __restrict__ hypwuP, const float* __restrict__ hypBm,
    float* __restrict__ logpx) {
  __shared__ __align__(16) float zb[4][16][16];
  __shared__ __align__(16) float ze[4][16][16];
  __shared__ __align__(16) float za[4][16][16];
  __shared__ __align__(16) f16   zcl[4][16][16];
  __shared__ __align__(16) f16   hhl[4][16][256];
  __shared__ __align__(16) float ccl[4][16][32];
  __shared__ float lpl[4][16], lal[4][16];
  __shared__ float cwzl[32][16];
  __shared__ float cbl[32];

  const int tid = threadIdx.x;
  const int w = tid >> 5, lane = tid & 31, lr = lane & 15, half = lane >> 4;
  const int sbase = blockIdx.x * 64;

  for (int e = tid; e < 64 * 16; e += 128) {
    int sm = e >> 4, i = e & 15;
    float zv = zin[(size_t)(sbase + sm) * 16 + i];
    zb[sm >> 4][sm & 15][i] = zv;
    ze[sm >> 4][sm & 15][i] = zv;
    za[sm >> 4][sm & 15][i] = 0.f;
  }
  for (int e = tid; e < 64 * 32; e += 128) {
    int sm = e >> 5, j = e & 31;
    ccl[sm >> 4][sm & 15][j] = cc[(size_t)(sbase + sm) * 32 + j];
  }
  for (int e = tid; e < 512; e += 128) cwzl[e >> 4][e & 15] = cwz[e];
  if (tid < 32) cbl[tid] = cbz[tid] + cbc[tid];
  if (tid < 64) { lpl[tid >> 4][tid & 15] = 0.f; lal[tid >> 4][tid & 15] = 0.f; }
  __syncthreads();   // only cross-wave sync needed; rest is intra-wave

  const float dt = -0.625f;
  const float s2 = 1.f + 0.5f * dt;
  const float s3 = 1.f + 0.5f * dt * s2;
  const float s4 = 1.f + dt * s3;
  const float F  = 1.f + dt * (1.f + dt * (0.5f + dt * (1.f / 6.f + dt / 24.f)));
  float cfac = 1.f;

  for (int step = 0; step < NSTEP; ++step) {
    for (int ev = 0; ev < 4; ++ev) {
      const int which = (ev == 0) ? 0 : ((ev == 3) ? 2 : 1);
      const int set = step * 3 + which;
      const float cs = cfac * (ev == 0 ? 1.f : (ev == 1 ? s2 : (ev == 2 ? s3 : s4)));

      // ---- gate: a = ze@cwz^T + (cbz+cbc) + cs*ccontrib ; zc = tanh*sigmoid
      float zev[16];
#pragma unroll
      for (int i = 0; i < 16; ++i) zev[i] = ze[w][lr][i];
#pragma unroll
      for (int j = 0; j < 16; ++j) {
        float a0 = cbl[j] + cs * ccl[w][lr][j];
        float a1 = cbl[j + 16] + cs * ccl[w][lr][j + 16];
#pragma unroll
        for (int i = 0; i < 16; ++i) {
          a0 += zev[i] * cwzl[j][i];
          a1 += zev[i] * cwzl[j + 16][i];
        }
        zcl[w][lr][j] = (f16)(tanhf(a0) * (1.f / (1.f + expf(-a1))));
      }

      // A fragment for zc (K=16 real, K=16..31 zero-padded)
      v16h afz;
      {
        v8h c0 = *(const v8h*)&zcl[w][lr][half * 8];
#pragma unroll
        for (int e = 0; e < 8; ++e) { afz[e] = c0[e]; afz[8 + e] = (f16)0.f; }
      }

      // ---- hh = tanh(zc @ Wm^T + Bm): 16 WMMA tiles of N=16
      const f16*  WmS = hypWmP + (size_t)set * WID * 32;   // [256][32] K-padded
      const float* BmS = hypBm + (size_t)set * WID;
      for (int nt = 0; nt < 16; ++nt) {
        v16h bf = load_b_frag(WmS + (size_t)(nt * 16 + lr) * 32, 0, half);
        v8f c = {};
        c = __builtin_amdgcn_wmma_f32_16x16x32_f16(false, afz, false, bf,
                                                   (short)0, c, false, false);
        float bn = BmS[nt * 16 + lr];
#pragma unroll
        for (int r = 0; r < 8; ++r) {
          int m = r + 8 * half;
          hhl[w][m][nt * 16 + lr] = (f16)tanhf(c[r] + bn);
        }
      }

      // ---- dz = hh @ Um / WID  and  trace = (1-hh^2) @ wu / WID
      const f16* UmTS = hypUmT + (size_t)set * WID * LD;
      const f16* wuS  = hypwuP + (size_t)set * LD * WID;   // [16][256], rows 1..15 = 0
      v8f cdz = {}, ctr = {};
      for (int k0 = 0; k0 < WID; k0 += 32) {
        v16h ah = load_a_frag(&hhl[w][lr][0], k0, half);
        v16h gh;
#pragma unroll
        for (int e = 0; e < 16; ++e) gh[e] = (f16)1.f - ah[e] * ah[e];
        v16h bu = load_b_frag(UmTS + (size_t)lr * WID, k0, half);
        v16h bw = load_b_frag(wuS + (size_t)lr * WID, k0, half);
        cdz = __builtin_amdgcn_wmma_f32_16x16x32_f16(false, ah, false, bu,
                                                     (short)0, cdz, false, false);
        ctr = __builtin_amdgcn_wmma_f32_16x16x32_f16(false, gh, false, bw,
                                                     (short)0, ctr, false, false);
      }

      // ---- RK4 bookkeeping
      const float wgt = (ev == 0 || ev == 3) ? 1.f : 2.f;
      const float cnext = (ev <= 1) ? dt * 0.5f : dt;
#pragma unroll
      for (int r = 0; r < 8; ++r) {
        int m = r + 8 * half;
        float dzv = cdz[r] * (1.f / (float)WID);
        float prev = (ev == 0) ? 0.f : za[w][m][lr];
        za[w][m][lr] = prev + wgt * dzv;
        if (ev < 3) ze[w][m][lr] = zb[w][m][lr] + cnext * dzv;
      }
      if (lr == 0) {   // trace lives in column 0 -> lanes 0 (m=0..7), 16 (m=8..15)
#pragma unroll
        for (int r = 0; r < 8; ++r) {
          int m = r + 8 * half;
          float lv = -ctr[r] * (1.f / (float)WID);
          float prev = (ev == 0) ? 0.f : lal[w][m];
          lal[w][m] = prev + wgt * lv;
        }
      }
      if (ev == 3) {
#pragma unroll
        for (int r = 0; r < 8; ++r) {
          int m = r + 8 * half;
          float nz = zb[w][m][lr] + (dt / 6.f) * za[w][m][lr];
          zb[w][m][lr] = nz;
          ze[w][m][lr] = nz;
        }
        if (lr == 0) {
#pragma unroll
          for (int r = 0; r < 8; ++r) {
            int m = r + 8 * half;
            lpl[w][m] += (dt / 6.f) * lal[w][m];
          }
        }
      }
    }
    cfac *= F;
  }

  // ---- logp_x = logp_z0 - logp_t0
  if (lane < 16) {
    float ss = 0.f;
#pragma unroll
    for (int i = 0; i < 16; ++i) { float v = zb[w][lane][i]; ss += v * v; }
    float lz0 = -0.5f * (16.f * 1.8378770664093453f + ss);
    logpx[sbase + w * 16 + lane] = lz0 - lpl[w][lane];
  }
}

// ---------------- deterministic mean reduction ----------------
__global__ __launch_bounds__(256) void mean_kernel(const float* __restrict__ lp,
                                                   float* __restrict__ out) {
  __shared__ float sh[256];
  int t = threadIdx.x;
  float s = 0.f;
  for (int i = t; i < B_SZ; i += 256) s += lp[i];
  sh[t] = s;
  __syncthreads();
  for (int o = 128; o > 0; o >>= 1) {
    if (t < o) sh[t] += sh[t + o];
    __syncthreads();
  }
  if (t == 0) out[0] = sh[0] / (float)B_SZ;
}

// ---------------------------------------------------------------------------
extern "C" void kernel_launch(void* const* d_in, const int* in_sizes, int n_in,
                              void* d_out, int out_size, void* d_ws, size_t ws_size,
                              hipStream_t stream) {
  (void)in_sizes; (void)n_in; (void)out_size; (void)ws_size;
  const float* input = (const float*)d_in[0];
  const float* eps   = (const float*)d_in[1];
  const float* ew0 = (const float*)d_in[2];  const float* eb0 = (const float*)d_in[3];
  const float* ew1 = (const float*)d_in[4];  const float* eb1 = (const float*)d_in[5];
  const float* ew2 = (const float*)d_in[6];  const float* eb2 = (const float*)d_in[7];
  const float* dw0 = (const float*)d_in[8];  const float* db0 = (const float*)d_in[9];
  const float* dw1 = (const float*)d_in[10]; const float* db1 = (const float*)d_in[11];
  const float* dw2 = (const float*)d_in[12]; const float* db2 = (const float*)d_in[13];
  const float* emb = (const float*)d_in[14];
  const float* cwz = (const float*)d_in[15]; const float* cbz = (const float*)d_in[16];
  const float* cwc = (const float*)d_in[17]; const float* cbc = (const float*)d_in[18];
  const float* hw1 = (const float*)d_in[19]; const float* hb1 = (const float*)d_in[20];
  const float* hw2 = (const float*)d_in[21]; const float* hb2 = (const float*)d_in[22];
  const float* hw3 = (const float*)d_in[23]; const float* hb3 = (const float*)d_in[24];
  const int*  cond = (const int*)d_in[25];

  float* out = (float*)d_out;
  float* out_recon  = out;                               // 8192*784
  float* out_scalar = out + (size_t)B_SZ * D_IN;         // 1
  float* out_mean   = out_scalar + 1;                    // 8192*16
  float* out_std    = out_mean + (size_t)B_SZ * LD;      // 8192*16

  char* ws = (char*)d_ws;
  size_t off = 0;
  auto take = [&](size_t bytes) -> void* {
    void* p = ws + off;
    off = (off + bytes + 255) & ~(size_t)255;
    return p;
  };
  f16*  X16   = (f16*)take((size_t)B_SZ * KP_IN * 2);
  f16*  EW0   = (f16*)take((size_t)HID * KP_IN * 2);
  f16*  EW1   = (f16*)take((size_t)HID * HID * 2);
  f16*  EW2   = (f16*)take((size_t)64 * HID * 2);        // 32 rows padded to 64
  f16*  DW0   = (f16*)take((size_t)HID * 32 * 2);
  f16*  DW1   = (f16*)take((size_t)HID * HID * 2);
  f16*  DW2   = (f16*)take((size_t)832 * HID * 2);       // 784 rows padded to 832
  f16*  H0    = (f16*)take((size_t)B_SZ * HID * 2);      // reused as decoder r0
  f16*  H1    = (f16*)take((size_t)B_SZ * HID * 2);      // reused as decoder r1
  float* MS   = (float*)take((size_t)B_SZ * 32 * 4);
  float* Z32  = (float*)take((size_t)B_SZ * LD * 4);
  f16*  Z16   = (f16*)take((size_t)B_SZ * 32 * 2);
  float* CC   = (float*)take((size_t)B_SZ * 32 * 4);
  f16*  WMP16 = (f16*)take((size_t)48 * WID * 32 * 2);   // [256][32] K-padded
  f16*  UMT16 = (f16*)take((size_t)48 * WID * LD * 2);
  f16*  WUP16 = (f16*)take((size_t)48 * LD * WID * 2);   // [16][256] row-padded
  float* BM32 = (float*)take((size_t)48 * WID * 4);
  float* LOGPX = (float*)take((size_t)B_SZ * 4);

  auto cvt = [&](const float* s, f16* d, int rows, int rowsPad, int K, int Kp) {
    long long total = (long long)rowsPad * Kp;
    convert_pad_kernel<<<dim3((unsigned)((total + 255) / 256)), dim3(256), 0, stream>>>(
        s, d, rows, rowsPad, K, Kp);
  };
  cvt(input, X16, B_SZ, B_SZ, D_IN, KP_IN);
  cvt(ew0, EW0, HID, HID, D_IN, KP_IN);
  cvt(ew1, EW1, HID, HID, HID, HID);
  cvt(ew2, EW2, 32, 64, HID, HID);
  cvt(dw0, DW0, HID, HID, LD, 32);
  cvt(dw1, DW1, HID, HID, HID, HID);
  cvt(dw2, DW2, D_IN, 832, HID, HID);

  // encoder
  gemm_nt_kernel<<<dim3(B_SZ / 32, 1), 256, 0, stream>>>(X16, EW0, eb0, KP_IN, HID, HID, 1,
                                                         H0, HID, nullptr, 0);
  gemm_nt_kernel<<<dim3(B_SZ / 32, 1), 256, 0, stream>>>(H0, EW1, eb1, HID, HID, HID, 2,
                                                         H1, HID, nullptr, 0);
  gemm_nt_kernel<<<dim3(B_SZ / 32, 1), 256, 0, stream>>>(H1, EW2, eb2, HID, 32, 64, 0,
                                                         nullptr, 0, MS, 32);
  latent_kernel<<<dim3(B_SZ * LD / 256), 256, 0, stream>>>(MS, eps, out_mean, out_std,
                                                           Z32, Z16);
  // decoder (H0/H1 reused)
  gemm_nt_kernel<<<dim3(B_SZ / 32, 1), 256, 0, stream>>>(Z16, DW0, db0, 32, HID, HID, 1,
                                                         H0, HID, nullptr, 0);
  gemm_nt_kernel<<<dim3(B_SZ / 32, 1), 256, 0, stream>>>(H0, DW1, db1, HID, HID, HID, 2,
                                                         H1, HID, nullptr, 0);
  gemm_nt_kernel<<<dim3(B_SZ / 32, 2), 256, 0, stream>>>(H1, DW2, db2, HID, D_IN, 832, 3,
                                                         nullptr, 0, out_recon, D_IN);
  // CNF
  ccond_kernel<<<dim3(B_SZ * 32 / 256), 256, 0, stream>>>(emb, cond, cwc, CC);
  hyper_kernel<<<dim3(48), 256, 0, stream>>>(hw1, hb1, hw2, hb2, hw3, hb3,
                                             WMP16, UMT16, WUP16, BM32);
  cnf_kernel<<<dim3(B_SZ / 64), 128, 0, stream>>>(Z32, CC, cwz, cbz, cbc,
                                                  WMP16, UMT16, WUP16, BM32, LOGPX);
  mean_kernel<<<dim3(1), 256, 0, stream>>>(LOGPX, out_scalar);
}
